// ModernAttention_16716012716444
// MI455X (gfx1250) — matrix operand order repaired and checked by reading, not compiled, not run on hardware
//
#include <hip/hip_runtime.h>
#include <hip/hip_bf16.h>
#include <stdint.h>
#include <stddef.h>

// Problem constants (from the reference)
#define B_    2
#define S_    2048
#define DIM_  1024
#define H_    16
#define HD_   64
#define HALF_ 32
#define NTOK  (B_ * S_)          // 4096
#define SCALE_ 0.125f            // HD^-0.5

typedef __attribute__((ext_vector_type(8)))  __bf16 v8bf;
typedef __attribute__((ext_vector_type(16))) __bf16 v16bf;
typedef __attribute__((ext_vector_type(8)))  float  v8f;
typedef __attribute__((ext_vector_type(4)))  int    v4i;

// ---------------------------------------------------------------------------
// Async global->LDS staging (gfx1250 path), guarded so compile never breaks.
// Probed signature: (v4i addrspace(1)*, v4i addrspace(3)*, imm, imm)
// ---------------------------------------------------------------------------
#if defined(__AMDGCN__) && __has_builtin(__builtin_amdgcn_global_load_async_to_lds_b128)
#define HAVE_ASYNC_LDS 1
#else
#define HAVE_ASYNC_LDS 0
#endif

__device__ __forceinline__ void copy16_g2l(const __bf16* g, __bf16* l) {
#if HAVE_ASYNC_LDS
  __builtin_amdgcn_global_load_async_to_lds_b128(
      (__attribute__((address_space(1))) v4i*)g,
      (__attribute__((address_space(3))) v4i*)l, 0, 0);
#else
  *(v8bf*)l = *(const v8bf*)g;
#endif
}

__device__ __forceinline__ void wait_stage() {
#if HAVE_ASYNC_LDS && __has_builtin(__builtin_amdgcn_s_wait_asynccnt)
  __builtin_amdgcn_s_wait_asynccnt(0);
#endif
}

// ---------------------------------------------------------------------------
// WMMA fragment loaders per CDNA5 ISA VGPR layouts (wave32).
// A (16x32 bf16): lane<16 -> row M=lane, K chunks {0..7, 16..23};
//                 lane>=16 -> same row, K chunks {8..15, 24..31}.
// ---------------------------------------------------------------------------
__device__ __forceinline__ v16bf frag_rm(const __bf16* base, int ld) {
  const int lane = threadIdx.x & 31;
  const int r = lane & 15, hi = lane >> 4;
  const __bf16* p = base + (size_t)r * ld + hi * 8;
  v8bf lo = *(const v8bf*)p;          // K = hi*8 .. hi*8+7
  v8bf hv = *(const v8bf*)(p + 16);   // K = 16+hi*8 .. 16+hi*8+7
  return __builtin_shufflevector(lo, hv, 0,1,2,3,4,5,6,7,8,9,10,11,12,13,14,15);
}

// B (32x16 bf16) from a row-major [N][K] source (row n contiguous in K):
// lane<16 -> col N=lane, K chunks {0..7, 16..23}; lane>=16 -> K chunks {8..15, 24..31}.
__device__ __forceinline__ v16bf fragB_rm(const __bf16* base, int ld) {
  const int lane = threadIdx.x & 31;
  const int nn = lane & 15, hi = lane >> 4;
  const __bf16* p = base + (size_t)nn * ld + hi * 8;
  v8bf lo = *(const v8bf*)p;
  v8bf hv = *(const v8bf*)(p + 16);
  return __builtin_shufflevector(lo, hv, 0,1,2,3,4,5,6,7,8,9,10,11,12,13,14,15);
}

__device__ __forceinline__ v8f wmma_bf16(v16bf a, v16bf b, v8f c) {
  return __builtin_amdgcn_wmma_f32_16x16x32_bf16(false, a, false, b, (short)0, c,
                                                 false, false);
}

__device__ __forceinline__ float redmax16(float v) {
#pragma unroll
  for (int m = 1; m <= 8; m <<= 1) v = fmaxf(v, __shfl_xor(v, m, 32));
  return v;
}
__device__ __forceinline__ float redsum16(float v) {
#pragma unroll
  for (int m = 1; m <= 8; m <<= 1) v += __shfl_xor(v, m, 32);
  return v;
}

// ---------------------------------------------------------------------------
// 1) fp32 -> bf16 conversion
// ---------------------------------------------------------------------------
__global__ void cvt_f32_bf16(const float* __restrict__ src, __bf16* __restrict__ dst,
                             int n) {
  for (int i = blockIdx.x * blockDim.x + threadIdx.x; i < n;
       i += gridDim.x * blockDim.x)
    dst[i] = (__bf16)src[i];
}

// ---------------------------------------------------------------------------
// 2/5) GEMM: out[m,n] = sum_k A[m,k] * W[n,k]  (A: M=4096 x K=1024, W: 1024x1024)
// Block tile 128x64, 256 threads (8 waves), each wave a 32x32 tile (2x2 WMMA).
// ---------------------------------------------------------------------------
template <bool OUTF32>
__global__ __launch_bounds__(256) void gemm_xwt(const __bf16* __restrict__ A,
                                                const __bf16* __restrict__ W,
                                                __bf16* __restrict__ outb,
                                                float* __restrict__ outf) {
  constexpr int Kd = DIM_;
  __shared__ __bf16 As[128][32];
  __shared__ __bf16 Ws[64][32];
  const int tid = threadIdx.x;
  const int wave = tid >> 5, lane = tid & 31;
  const int nn = lane & 15, mb = (lane >> 4) * 8;
  const int blockM = blockIdx.y * 128, blockN = blockIdx.x * 64;
  const int waveM = (wave >> 1) * 32, waveN = (wave & 1) * 32;

  v8f acc[4];
#pragma unroll
  for (int t = 0; t < 4; ++t) acc[t] = (v8f){0, 0, 0, 0, 0, 0, 0, 0};

  for (int k0 = 0; k0 < Kd; k0 += 32) {
    __syncthreads();
    for (int i = tid; i < 512; i += 256) {           // A tile: 128 rows x 32 cols
      int row = i >> 2, ch = i & 3;
      copy16_g2l(A + (size_t)(blockM + row) * Kd + k0 + ch * 8, &As[row][ch * 8]);
    }
    {                                                // W tile: 64 rows x 32 cols
      int row = tid >> 2, ch = tid & 3;
      copy16_g2l(W + (size_t)(blockN + row) * Kd + k0 + ch * 8, &Ws[row][ch * 8]);
    }
    if (k0 + 32 < Kd)                                // global_prefetch_b8
      __builtin_prefetch(A + (size_t)(blockM + (tid >> 1)) * Kd + k0 + 32 +
                             (tid & 1) * 16, 0, 1);
    wait_stage();
    __syncthreads();

    v16bf a0 = frag_rm(&As[waveM][0], 32);
    v16bf a1 = frag_rm(&As[waveM + 16][0], 32);
    v16bf b0 = fragB_rm(&Ws[waveN][0], 32);
    v16bf b1 = fragB_rm(&Ws[waveN + 16][0], 32);
    acc[0] = wmma_bf16(a0, b0, acc[0]);
    acc[1] = wmma_bf16(a0, b1, acc[1]);
    acc[2] = wmma_bf16(a1, b0, acc[2]);
    acc[3] = wmma_bf16(a1, b1, acc[3]);
  }

#pragma unroll
  for (int t = 0; t < 4; ++t) {
    const int ti = t >> 1, tj = t & 1;
#pragma unroll
    for (int r = 0; r < 8; ++r) {
      const size_t row = (size_t)blockM + waveM + ti * 16 + mb + r;
      const size_t col = (size_t)blockN + waveN + tj * 16 + nn;
      if (OUTF32)
        outf[row * DIM_ + col] = acc[t][r];
      else
        outb[row * DIM_ + col] = (__bf16)acc[t][r];
    }
  }
}

// ---------------------------------------------------------------------------
// 3) RoPE on Q and K (bf16 in place, fp32 math)
// ---------------------------------------------------------------------------
__global__ void rope_kernel(__bf16* __restrict__ q, __bf16* __restrict__ kk,
                            const float* __restrict__ cosv,
                            const float* __restrict__ sinv) {
  const int total = NTOK * H_ * HALF_;
  int idx = blockIdx.x * blockDim.x + threadIdx.x;
  if (idx >= 2 * total) return;
  __bf16* t = (idx < total) ? q : kk;
  int rem = (idx < total) ? idx : idx - total;
  int m = rem / (H_ * HALF_);
  int r2 = rem - m * (H_ * HALF_);
  int h = r2 / HALF_;
  int i = r2 - h * HALF_;
  int s = m & (S_ - 1);
  __bf16* base = t + (size_t)m * DIM_ + h * HD_ + 2 * i;
  float c = cosv[s * HALF_ + i], sn = sinv[s * HALF_ + i];
  float e = (float)base[0], o = (float)base[1];
  base[0] = (__bf16)(e * c - o * sn);
  base[1] = (__bf16)(e * sn + o * c);
}

// ---------------------------------------------------------------------------
// 4) Causal flash attention. Grid (S/64, H, B), 128 threads (4 waves).
// Each wave owns a 16-row Q strip; K staged async (raw), V staged TRANSPOSED
// (Vt[d][key]) so every P*V B-fragment is a contiguous ds_load_b128.
// ---------------------------------------------------------------------------
__global__ __launch_bounds__(128) void flash_kernel(const __bf16* __restrict__ q,
                                                    const __bf16* __restrict__ k,
                                                    const __bf16* __restrict__ v,
                                                    __bf16* __restrict__ o) {
  __shared__ __bf16 Ks[32][64];   // [key][d]
  __shared__ __bf16 Vt[64][32];   // [d][key]  (transposed V chunk)
  __shared__ __bf16 Ps[4][16][32];
  const int tid = threadIdx.x, wave = tid >> 5, lane = tid & 31;
  const int nn = lane & 15, mb = (lane >> 4) * 8;
  const int qs = blockIdx.x * 64, h = blockIdx.y, b = blockIdx.z;
  const size_t qrow0 = (size_t)b * S_ + qs + wave * 16;

  const __bf16* qbase = q + qrow0 * DIM_ + h * HD_;
  const v16bf Qf0 = frag_rm(qbase, DIM_);        // d = 0..31
  const v16bf Qf1 = frag_rm(qbase + 32, DIM_);   // d = 32..63

  v8f o0 = {0,0,0,0,0,0,0,0}, o1 = {0,0,0,0,0,0,0,0};
  v8f o2 = {0,0,0,0,0,0,0,0}, o3 = {0,0,0,0,0,0,0,0};
  float mrow[8], lrow[8];
#pragma unroll
  for (int r = 0; r < 8; ++r) { mrow[r] = -3.0e38f; lrow[r] = 0.0f; }

  const size_t kvbase = (size_t)b * S_ * DIM_ + h * HD_;
  const int nchunk = qs / 32 + 2;   // causal: keys 0 .. qs+63, uniform per block

  for (int c = 0; c < nchunk; ++c) {
    const int kc = c * 32;
    __syncthreads();
    // K chunk (32 keys x 64 d): async global->LDS, row-major
    for (int i = tid; i < 256; i += 128) {
      int row = i >> 3, ch = i & 7;
      copy16_g2l(k + kvbase + (size_t)(kc + row) * DIM_ + ch * 8, &Ks[row][ch * 8]);
    }
    // V chunk: load 8 d-values per key to registers, store transposed to Vt
    for (int i = tid; i < 256; i += 128) {
      int key = i >> 3, dch = i & 7;
      v8bf val = *(const v8bf*)(v + kvbase + (size_t)(kc + key) * DIM_ + dch * 8);
#pragma unroll
      for (int e = 0; e < 8; ++e) Vt[dch * 8 + e][key] = val[e];
    }
    wait_stage();
    __syncthreads();

    // scores: two 16x16 tiles for this wave's 16 q-rows (hoist all K frags)
    const v16bf kf00 = fragB_rm(&Ks[0][0], 64);
    const v16bf kf01 = fragB_rm(&Ks[0][32], 64);
    const v16bf kf10 = fragB_rm(&Ks[16][0], 64);
    const v16bf kf11 = fragB_rm(&Ks[16][32], 64);
    v8f s0 = {0,0,0,0,0,0,0,0}, s1 = {0,0,0,0,0,0,0,0};
    s0 = wmma_bf16(Qf0, kf00, s0);
    s0 = wmma_bf16(Qf1, kf01, s0);
    s1 = wmma_bf16(Qf0, kf10, s1);
    s1 = wmma_bf16(Qf1, kf11, s1);

    float sc[8];
#pragma unroll
    for (int r = 0; r < 8; ++r) {
      const int qrow = qs + wave * 16 + mb + r;
      float a = s0[r] * SCALE_;  if (kc + nn > qrow)      a = -3.0e38f;
      float bb = s1[r] * SCALE_; if (kc + 16 + nn > qrow) bb = -3.0e38f;
      const float mx = redmax16(fmaxf(a, bb));
      const float mnew = fmaxf(mrow[r], mx);
      const float co = __expf(mrow[r] - mnew);
      const float p0 = __expf(a - mnew);
      const float p1 = __expf(bb - mnew);
      lrow[r] = lrow[r] * co + redsum16(p0 + p1);
      mrow[r] = mnew;
      sc[r] = co;
      Ps[wave][mb + r][nn]      = (__bf16)p0;
      Ps[wave][mb + r][16 + nn] = (__bf16)p1;
    }
    __syncthreads();

#pragma unroll
    for (int r = 0; r < 8; ++r) {
      o0[r] *= sc[r]; o1[r] *= sc[r]; o2[r] *= sc[r]; o3[r] *= sc[r];
    }
    const v16bf Pf = frag_rm(&Ps[wave][0][0], 32);   // A fragment: 16 q x 32 keys
    o0 = wmma_bf16(Pf, fragB_rm(&Vt[0][0],  32), o0);
    o1 = wmma_bf16(Pf, fragB_rm(&Vt[16][0], 32), o1);
    o2 = wmma_bf16(Pf, fragB_rm(&Vt[32][0], 32), o2);
    o3 = wmma_bf16(Pf, fragB_rm(&Vt[48][0], 32), o3);
  }

#pragma unroll
  for (int r = 0; r < 8; ++r) {
    const size_t row = qrow0 + mb + r;
    const float inv = 1.0f / lrow[r];
    __bf16* op = o + row * DIM_ + h * HD_ + nn;
    op[0]  = (__bf16)(o0[r] * inv);
    op[16] = (__bf16)(o1[r] * inv);
    op[32] = (__bf16)(o2[r] * inv);
    op[48] = (__bf16)(o3[r] * inv);
  }
}

// ---------------------------------------------------------------------------
// Orchestration
// ---------------------------------------------------------------------------
extern "C" void kernel_launch(void* const* d_in, const int* in_sizes, int n_in,
                              void* d_out, int out_size, void* d_ws, size_t ws_size,
                              hipStream_t stream) {
  const float* x    = (const float*)d_in[0];
  const float* cosv = (const float*)d_in[1];
  const float* sinv = (const float*)d_in[2];
  // d_in[3] = mask (int), unused: causal mask computed analytically
  const float* wq = (const float*)d_in[4];
  const float* wk = (const float*)d_in[5];
  const float* wv = (const float*)d_in[6];
  const float* wo = (const float*)d_in[7];
  float* out = (float*)d_out;

  char* ws = (char*)d_ws;
  const size_t MB = 1u << 20;
  __bf16* xb  = (__bf16*)(ws + 0);        // 8 MB : x in bf16
  __bf16* wqb = (__bf16*)(ws + 8  * MB);  // 2 MB each
  __bf16* wkb = (__bf16*)(ws + 10 * MB);
  __bf16* wvb = (__bf16*)(ws + 12 * MB);
  __bf16* wob = (__bf16*)(ws + 14 * MB);
  __bf16* qb  = (__bf16*)(ws + 16 * MB);  // 8 MB
  __bf16* kb  = (__bf16*)(ws + 24 * MB);  // 8 MB
  __bf16* vb  = (__bf16*)(ws + 32 * MB);  // 8 MB
  __bf16* ab  = (__bf16*)(ws + 40 * MB);  // 8 MB attention output (pre-proj)

  const int nx = NTOK * DIM_;     // 4194304
  const int nw = DIM_ * DIM_;     // 1048576
  cvt_f32_bf16<<<nx / 256, 256, 0, stream>>>(x, xb, nx);
  cvt_f32_bf16<<<nw / 256, 256, 0, stream>>>(wq, wqb, nw);
  cvt_f32_bf16<<<nw / 256, 256, 0, stream>>>(wk, wkb, nw);
  cvt_f32_bf16<<<nw / 256, 256, 0, stream>>>(wv, wvb, nw);
  cvt_f32_bf16<<<nw / 256, 256, 0, stream>>>(wo, wob, nw);

  dim3 ggrid(DIM_ / 64, NTOK / 128);  // (16, 32)
  gemm_xwt<false><<<ggrid, 256, 0, stream>>>(xb, wqb, qb, nullptr);
  gemm_xwt<false><<<ggrid, 256, 0, stream>>>(xb, wkb, kb, nullptr);
  gemm_xwt<false><<<ggrid, 256, 0, stream>>>(xb, wvb, vb, nullptr);

  const int nrope = 2 * NTOK * H_ * HALF_;  // 4194304
  rope_kernel<<<nrope / 256, 256, 0, stream>>>(qb, kb, cosv, sinv);

  dim3 fgrid(S_ / 64, H_, B_);  // (32, 16, 2)
  flash_kernel<<<fgrid, 128, 0, stream>>>(qb, kb, vb, ab);

  gemm_xwt<true><<<ggrid, 256, 0, stream>>>(ab, wob, nullptr, out);
}
